// FusedSideAttention_79319456023304
// MI455X (gfx1250) — matrix operand order repaired and checked by reading, not compiled, not run on hardware
//
#include <hip/hip_runtime.h>
#include <cstddef>

typedef __bf16 bf16_t;
typedef __attribute__((ext_vector_type(16))) bf16_t v16bf;
typedef __attribute__((ext_vector_type(8)))  bf16_t v8bf;
typedef __attribute__((ext_vector_type(8)))  float  v8f;

union ABu { v16bf v; v8bf h[2]; };

#define DIM   1024
#define HEADS 16
#define HD    64
#define QLEN  2048
#define SLEN  512
#define KLEN  2560   // QLEN + SLEN
#define BATCH 2
#define MAIN_ROWS 4096   // BATCH*QLEN
#define KV_ROWS   4608   // MAIN_ROWS + SLEN

__device__ __forceinline__ v8f wmma_bf16(v16bf a, v16bf b, v8f c) {
  // D = A(16x32 bf16) * B(32x16 bf16) + C(16x16 f32)
  return __builtin_amdgcn_wmma_f32_16x16x32_bf16(
      /*neg_a=*/false, a, /*neg_b=*/false, b,
      /*c_mod=*/(short)0, c, /*reuse_a=*/false, /*reuse_b=*/false);
}

// Async global->LDS copy, 16 bytes per lane (CDNA5 path, tracked by ASYNCcnt).
__device__ __forceinline__ void async_load_b128(unsigned lds_byte_addr,
                                                const void* gptr) {
  asm volatile("global_load_async_to_lds_b128 %0, %1, off"
               :: "v"(lds_byte_addr), "v"((unsigned long long)(size_t)gptr)
               : "memory");
}
__device__ __forceinline__ void wait_async0() {
  asm volatile("s_wait_asynccnt 0" ::: "memory");
}
__device__ __forceinline__ unsigned lds_addr_of(const void* p) {
  // Low 32 bits of a flat shared-aperture address are the LDS byte offset.
  return (unsigned)(size_t)p;
}

// ---------------------------------------------------------------------------
// Prep: concat(main, side) rows -> bf16  [KV_ROWS][DIM]
// ---------------------------------------------------------------------------
__global__ void convert_rows_kernel(const float* __restrict__ main_in,
                                    const float* __restrict__ side_in,
                                    bf16_t* __restrict__ xcat) {
  int idx = blockIdx.x * blockDim.x + threadIdx.x;   // over KV_ROWS*DIM
  int r = idx >> 10, c = idx & (DIM - 1);
  float v = (r < MAIN_ROWS) ? main_in[idx]
                            : side_in[(r - MAIN_ROWS) * DIM + c];
  xcat[idx] = (bf16_t)v;
}

// ---------------------------------------------------------------------------
// Prep: Wt[n][k] = (bf16)W[k][n], LDS-tiled transpose (coalesced both sides)
// ---------------------------------------------------------------------------
__global__ void transpose_w_kernel(const float* __restrict__ W,
                                   bf16_t* __restrict__ Wt) {
  __shared__ float tile[32][33];
  int bx = blockIdx.x & 31, by = blockIdx.x >> 5;
  int tx = threadIdx.x & 31, ty = threadIdx.x >> 5;   // 256 thr = 32x8
  #pragma unroll
  for (int i = 0; i < 32; i += 8)
    tile[ty + i][tx] = W[(by * 32 + ty + i) * DIM + bx * 32 + tx];
  __syncthreads();
  #pragma unroll
  for (int i = 0; i < 32; i += 8)
    Wt[(bx * 32 + ty + i) * DIM + by * 32 + tx] = (bf16_t)tile[tx][ty + i];
}

// ---------------------------------------------------------------------------
// WMMA GEMM: Y[rows,DIM] = X[rows,DIM](bf16) @ Wt^T(bf16) + bias
// Block = 8 waves: 128 rows x 64 cols. The 64x32 Wt tile is async-staged into
// double-buffered LDS and shared by all waves; A-fragments are register
// double-buffered. One wave -> 16x64 tile (4 accumulators), K-step 32.
// mode 0: scatter to qh [B,H,QLEN,64]
// mode 1: scatter to kh [B,H,KLEN,64]   (side rows duplicated per batch)
// mode 2: scatter to vT [B,H,64,KLEN]   (transposed V)
// mode 3: f32 store to out [rows,DIM]
// ---------------------------------------------------------------------------
__global__ void proj_gemm_kernel(const bf16_t* __restrict__ X,
                                 const bf16_t* __restrict__ Wt,
                                 const float*  __restrict__ bias,
                                 int mode,
                                 bf16_t* __restrict__ out_q,
                                 bf16_t* __restrict__ out_k,
                                 bf16_t* __restrict__ out_vT,
                                 float*  __restrict__ out_f32) {
  __shared__ __align__(32) bf16_t btile[2][64 * 32];   // [col][k], 4KB each

  const int tid  = threadIdx.x;
  const int lane = tid & 31;
  const int wave = tid >> 5;
  const int rowg  = blockIdx.x >> 4;        // 128-row group
  const int nbase = (blockIdx.x & 15) * 64; // 64-col group
  const int n16  = lane & 15;
  const int half = lane >> 4;

  // async staging: thread -> 16B of the 64x32 Wt tile
  const int srow = tid >> 2;                // 0..63 : Wt row (output col)
  const int scol = (tid & 3) * 8;           // element offset inside 32-K chunk
  const bf16_t* wsrc = Wt + (size_t)(nbase + srow) * DIM + scol;
  const unsigned ldsdst0 = lds_addr_of(&btile[0][srow * 32 + scol]);
  const unsigned ldsdst1 = lds_addr_of(&btile[1][srow * 32 + scol]);

  v8f acc0 = {}, acc1 = {}, acc2 = {}, acc3 = {};
  v8f* acc[4] = {&acc0, &acc1, &acc2, &acc3};

  const bf16_t* ap = X + (size_t)(rowg * 128 + wave * 16 + n16) * DIM;

  // prologue: stage k0=0 tile, prefetch first A fragment
  async_load_b128(ldsdst0, wsrc);
  ABu a_cur, a_nxt;
  a_cur.h[0] = *(const v8bf*)(ap + 16 * half);
  a_cur.h[1] = *(const v8bf*)(ap + 8 + 16 * half);
  a_nxt = a_cur;
  wait_async0();
  __syncthreads();

  for (int k0 = 0; k0 < DIM; k0 += 32) {
    const int cur = (k0 >> 5) & 1;
    if (k0 + 32 < DIM) {
      async_load_b128(cur ? ldsdst0 : ldsdst1, wsrc + k0 + 32);
      a_nxt.h[0] = *(const v8bf*)(ap + k0 + 32 + 16 * half);
      a_nxt.h[1] = *(const v8bf*)(ap + k0 + 40 + 16 * half);
    }
    #pragma unroll
    for (int t = 0; t < 4; ++t) {
      const bf16_t* bp = &btile[cur][(t * 16 + n16) * 32 + 16 * half];
      *acc[t] = wmma_bf16(a_cur.v, *(const v16bf*)bp, *acc[t]);
    }
    a_cur = a_nxt;
    wait_async0();     // next tile landed in the other buffer
    __syncthreads();   // all waves done reading current buffer
  }

  #pragma unroll
  for (int t = 0; t < 4; ++t) {
    const int n  = nbase + t * 16 + n16;
    const float bv = bias[n];
    const int h  = n >> 6, dd = n & 63;
    #pragma unroll
    for (int r = 0; r < 8; ++r) {
      const int g = rowg * 128 + wave * 16 + r + 8 * half;  // global row
      const float val = (*acc[t])[r] + bv;
      if (mode == 3) {
        out_f32[(size_t)g * DIM + n] = val;
      } else {
        const bf16_t bb = (bf16_t)val;
        if (mode == 0) {
          const int b = g >> 11, qi = g & (QLEN - 1);
          out_q[(((size_t)(b * HEADS + h)) * QLEN + qi) * HD + dd] = bb;
        } else if (mode == 1) {
          if (g < MAIN_ROWS) {
            const int b = g >> 11, key = g & (QLEN - 1);
            out_k[(((size_t)(b * HEADS + h)) * KLEN + key) * HD + dd] = bb;
          } else {
            const int key = QLEN + (g - MAIN_ROWS);
            out_k[(((size_t)h) * KLEN + key) * HD + dd] = bb;
            out_k[(((size_t)(HEADS + h)) * KLEN + key) * HD + dd] = bb;
          }
        } else {  // mode 2: transposed V
          if (g < MAIN_ROWS) {
            const int b = g >> 11, key = g & (QLEN - 1);
            out_vT[(((size_t)(b * HEADS + h)) * HD + dd) * KLEN + key] = bb;
          } else {
            const int key = QLEN + (g - MAIN_ROWS);
            out_vT[(((size_t)h) * HD + dd) * KLEN + key] = bb;
            out_vT[(((size_t)(HEADS + h)) * HD + dd) * KLEN + key] = bb;
          }
        }
      }
    }
  }
}

// ---------------------------------------------------------------------------
// Flash attention, block-cooperative: 8 waves share (b,h) over 128 queries.
// Per 32-key chunk, K (32x64) and V^T (64x32) tiles are async-staged into
// double-buffered LDS shared by all waves; online softmax uses 16-lane
// shuffle reductions; P relayout C->A goes through a per-wave LDS buffer.
// ---------------------------------------------------------------------------
__global__ void attn_kernel(const bf16_t* __restrict__ qh,
                            const bf16_t* __restrict__ kh,
                            const bf16_t* __restrict__ vT,
                            const int*    __restrict__ mask,
                            bf16_t* __restrict__ att) {
  __shared__ __align__(32) bf16_t ktile[2][32 * 64];  // [key][dim]  4KB each
  __shared__ __align__(32) bf16_t vtile[2][64 * 32];  // [dim][key]  4KB each
  __shared__ __align__(32) bf16_t pbuf[8][16 * 32];   // per-wave P tile

  const int tid  = threadIdx.x;
  const int lane = tid & 31;
  const int wave = tid >> 5;
  const int qg = blockIdx.x & 15;           // 128-query group
  const int h  = (blockIdx.x >> 4) & 15;
  const int b  = blockIdx.x >> 8;
  const int qbase = qg * 128 + wave * 16;
  const int n16  = lane & 15;
  const int half = lane >> 4;
  const size_t bh = (size_t)(b * HEADS + h);

  // async staging: thread -> 16B of each tile
  const int krow = tid >> 3;                // 0..31 key
  const int kcol = (tid & 7) * 8;           // dim element offset
  const bf16_t* ksrc = kh + (bh * KLEN + krow) * HD + kcol;
  const unsigned kdst0 = lds_addr_of(&ktile[0][krow * 64 + kcol]);
  const unsigned kdst1 = lds_addr_of(&ktile[1][krow * 64 + kcol]);
  const int vrow = tid >> 2;                // 0..63 dim
  const int vcol = (tid & 3) * 8;           // key element offset
  const bf16_t* vsrc = vT + (bh * HD + vrow) * KLEN + vcol;
  const unsigned vdst0 = lds_addr_of(&vtile[0][vrow * 32 + vcol]);
  const unsigned vdst1 = lds_addr_of(&vtile[1][vrow * 32 + vcol]);

  // Q A-fragments (K = dims 0..31 and 32..63), resident for whole loop
  ABu aq0, aq1;
  {
    const bf16_t* ap = qh + (bh * QLEN + qbase + n16) * HD;
    aq0.h[0] = *(const v8bf*)(ap + 16 * half);
    aq0.h[1] = *(const v8bf*)(ap + 8 + 16 * half);
    aq1.h[0] = *(const v8bf*)(ap + 32 + 16 * half);
    aq1.h[1] = *(const v8bf*)(ap + 40 + 16 * half);
  }

  float mrow[8], lrow[8];
  #pragma unroll
  for (int r = 0; r < 8; ++r) { mrow[r] = -3.0e38f; lrow[r] = 0.0f; }
  v8f acc0 = {}, acc1 = {}, acc2 = {}, acc3 = {};
  v8f* acc[4] = {&acc0, &acc1, &acc2, &acc3};

  bf16_t* pb = &pbuf[wave][0];

  // prologue: stage chunk 0
  async_load_b128(kdst0, ksrc);
  async_load_b128(vdst0, vsrc);
  wait_async0();
  __syncthreads();

  for (int j0 = 0; j0 < KLEN; j0 += 32) {
    const int cur = (j0 >> 5) & 1;
    if (j0 + 32 < KLEN) {   // stage next chunk into the other buffer
      async_load_b128(cur ? kdst0 : kdst1, ksrc + (size_t)(j0 + 32) * HD);
      async_load_b128(cur ? vdst0 : vdst1, vsrc + (j0 + 32));
    }

    // ---- scores: two 16-key tiles from LDS K tile ----
    v8f s0, s1;
    {
      const bf16_t* kp0 = &ktile[cur][n16 * 64 + 16 * half];
      v8f z = {};
      z  = wmma_bf16(aq0.v, *(const v16bf*)(kp0),      z);
      s0 = wmma_bf16(aq1.v, *(const v16bf*)(kp0 + 32), z);
      const bf16_t* kp1 = kp0 + 16 * 64;
      v8f z1 = {};
      z1 = wmma_bf16(aq0.v, *(const v16bf*)(kp1),      z1);
      s1 = wmma_bf16(aq1.v, *(const v16bf*)(kp1 + 32), z1);
    }

    // ---- scale + mask ----
    const int* mp = mask + ((size_t)b * QLEN + qbase + 8 * half) * KLEN + j0 + n16;
    #pragma unroll
    for (int r = 0; r < 8; ++r) {
      const int m0 = mp[(size_t)r * KLEN];
      const int m1 = mp[(size_t)r * KLEN + 16];
      s0[r] = s0[r] * 0.125f + (m0 ? 0.0f : -1.0e9f);
      s1[r] = s1[r] * 0.125f + (m1 ? 0.0f : -1.0e9f);
    }

    // ---- online softmax (16-lane row reductions via shfl_xor) ----
    #pragma unroll
    for (int r = 0; r < 8; ++r) {
      float cm = fmaxf(s0[r], s1[r]);
      cm = fmaxf(cm, __shfl_xor(cm, 1, 32));
      cm = fmaxf(cm, __shfl_xor(cm, 2, 32));
      cm = fmaxf(cm, __shfl_xor(cm, 4, 32));
      cm = fmaxf(cm, __shfl_xor(cm, 8, 32));
      const float mnew  = fmaxf(mrow[r], cm);
      const float alpha = __expf(mrow[r] - mnew);
      mrow[r] = mnew;
      const float p0 = __expf(s0[r] - mnew);
      const float p1 = __expf(s1[r] - mnew);
      s0[r] = p0; s1[r] = p1;
      float rs = p0 + p1;
      rs += __shfl_xor(rs, 1, 32);
      rs += __shfl_xor(rs, 2, 32);
      rs += __shfl_xor(rs, 4, 32);
      rs += __shfl_xor(rs, 8, 32);
      lrow[r] = lrow[r] * alpha + rs;
      (*acc[0])[r] *= alpha; (*acc[1])[r] *= alpha;
      (*acc[2])[r] *= alpha; (*acc[3])[r] *= alpha;
    }

    // ---- P: C-layout -> LDS -> A-layout ----
    #pragma unroll
    for (int r = 0; r < 8; ++r) {
      pb[(r + 8 * half) * 32 + n16]      = (bf16_t)s0[r];
      pb[(r + 8 * half) * 32 + 16 + n16] = (bf16_t)s1[r];
    }
    asm volatile("s_wait_dscnt 0" ::: "memory");
    ABu pa;
    {
      const bf16_t* pr = pb + n16 * 32;
      pa.h[0] = *(const v8bf*)(pr + 16 * half);
      pa.h[1] = *(const v8bf*)(pr + 8 + 16 * half);
    }

    // ---- P @ V : 4 d-column tiles from LDS V tile ----
    #pragma unroll
    for (int t = 0; t < 4; ++t) {
      const bf16_t* vp = &vtile[cur][(t * 16 + n16) * 32 + 16 * half];
      *acc[t] = wmma_bf16(pa.v, *(const v16bf*)vp, *acc[t]);
    }

    wait_async0();     // next chunk landed in the other buffers
    __syncthreads();   // all waves done reading current buffers
  }

  // ---- normalize and emit bf16 att [B,Q,DIM] ----
  #pragma unroll
  for (int t = 0; t < 4; ++t) {
    #pragma unroll
    for (int r = 0; r < 8; ++r) {
      const float val = (*acc[t])[r] / lrow[r];
      const int row = qbase + r + 8 * half;
      att[((size_t)b * QLEN + row) * DIM + h * HD + t * 16 + n16] = (bf16_t)val;
    }
  }
}

// ---------------------------------------------------------------------------
// Host driver
// ---------------------------------------------------------------------------
extern "C" void kernel_launch(void* const* d_in, const int* in_sizes, int n_in,
                              void* d_out, int out_size, void* d_ws, size_t ws_size,
                              hipStream_t stream) {
  (void)in_sizes; (void)n_in; (void)out_size; (void)ws_size;
  const float* main_in = (const float*)d_in[0];
  const float* side_in = (const float*)d_in[1];
  const int*   mask    = (const int*)  d_in[2];
  const float* Wq = (const float*)d_in[3];
  const float* bq = (const float*)d_in[4];
  const float* Wk = (const float*)d_in[5];
  const float* bk = (const float*)d_in[6];
  const float* Wv = (const float*)d_in[7];
  const float* bv = (const float*)d_in[8];
  const float* Wo = (const float*)d_in[9];
  const float* bo = (const float*)d_in[10];
  float* out = (float*)d_out;

  char* ws = (char*)d_ws;
  size_t off = 0;
  auto alloc = [&](size_t bytes) -> char* {
    char* p = ws + off;
    off += (bytes + 255) & ~(size_t)255;
    return p;
  };
  bf16_t* xcat = (bf16_t*)alloc((size_t)KV_ROWS * DIM * 2);
  bf16_t* Wtq  = (bf16_t*)alloc((size_t)DIM * DIM * 2);
  bf16_t* Wtk  = (bf16_t*)alloc((size_t)DIM * DIM * 2);
  bf16_t* Wtv  = (bf16_t*)alloc((size_t)DIM * DIM * 2);
  bf16_t* Wto  = (bf16_t*)alloc((size_t)DIM * DIM * 2);
  bf16_t* qh   = (bf16_t*)alloc((size_t)BATCH * HEADS * QLEN * HD * 2);
  bf16_t* kh   = (bf16_t*)alloc((size_t)BATCH * HEADS * KLEN * HD * 2);
  bf16_t* vT   = (bf16_t*)alloc((size_t)BATCH * HEADS * HD * KLEN * 2);
  bf16_t* att  = (bf16_t*)alloc((size_t)MAIN_ROWS * DIM * 2);

  // prep
  convert_rows_kernel<<<(KV_ROWS * DIM) / 256, 256, 0, stream>>>(main_in, side_in, xcat);
  transpose_w_kernel<<<1024, 256, 0, stream>>>(Wq, Wtq);
  transpose_w_kernel<<<1024, 256, 0, stream>>>(Wk, Wtk);
  transpose_w_kernel<<<1024, 256, 0, stream>>>(Wv, Wtv);
  transpose_w_kernel<<<1024, 256, 0, stream>>>(Wo, Wto);

  // QKV projections (WMMA, async-LDS double buffered)
  proj_gemm_kernel<<<(MAIN_ROWS / 128) * 16, 256, 0, stream>>>(
      xcat, Wtq, bq, 0, qh, nullptr, nullptr, nullptr);
  proj_gemm_kernel<<<(KV_ROWS / 128) * 16, 256, 0, stream>>>(
      xcat, Wtk, bk, 1, nullptr, kh, nullptr, nullptr);
  proj_gemm_kernel<<<(KV_ROWS / 128) * 16, 256, 0, stream>>>(
      xcat, Wtv, bv, 2, nullptr, nullptr, vT, nullptr);

  // flash attention (WMMA scores + WMMA P@V, shared async K/V tiles)
  attn_kernel<<<BATCH * HEADS * (QLEN / 128), 256, 0, stream>>>(
      qh, kh, vT, mask, att);

  // output projection (WMMA, f32 out + bias)
  proj_gemm_kernel<<<(MAIN_ROWS / 128) * 16, 256, 0, stream>>>(
      att, Wto, bo, 3, nullptr, nullptr, nullptr, out);
}
